// GATTransformerEncoderDecoder_44641890075100
// MI455X (gfx1250) — compile-verified, matrix operand order checked by reading, not run on hardware
//
#include <hip/hip_runtime.h>
#include <hip/hip_bf16.h>

// ---------------- problem constants (from reference) ----------------
constexpr int N_NODES = 20000;
constexpr int E_EDGES = 320000;
constexpr int ETOT    = E_EDGES + N_NODES;   // PyG adds self-loops
constexpr int F_IN    = 16;
constexpr int F_PAD   = 32;                  // K padded to WMMA depth
constexpr int D_FEAT  = 64;
constexpr int H_HEADS = 4;
constexpr int C_CH    = 64;
constexpr int HC      = 256;                 // H*C
constexpr int DM      = 256;
constexpr int FFD     = 1024;
constexpr int OUTD    = 128;
constexpr int G_GRAPHS= 32;

typedef __attribute__((ext_vector_type(16))) __bf16 v16bf;
typedef __attribute__((ext_vector_type(8)))  float  v8f;
typedef unsigned int uint32x4 __attribute__((ext_vector_type(4)));
typedef int          int32x8  __attribute__((ext_vector_type(8)));
typedef int          int32x4  __attribute__((ext_vector_type(4)));

__device__ __forceinline__ unsigned short f2bf(float f) {
  union { float f; unsigned u; } x; x.f = f;
  unsigned u = x.u;
  return (unsigned short)((u + 0x7FFFu + ((u >> 16) & 1u)) >> 16);
}

// order-preserving float<->uint mapping for atomicMax-based segment max
__device__ __forceinline__ unsigned fkey(float f) {
  unsigned u = __float_as_uint(f);
  return (u & 0x80000000u) ? ~u : (u | 0x80000000u);
}
__device__ __forceinline__ float fkey_inv(unsigned k) {
  unsigned u = (k & 0x80000000u) ? (k & 0x7FFFFFFFu) : ~k;
  return __uint_as_float(u);
}

// ---------------- elementwise helpers ----------------
__global__ void cvt_bf16_kernel(const float* __restrict__ in,
                                unsigned short* __restrict__ out, int n) {
  int i = blockIdx.x * blockDim.x + threadIdx.x;
  if (i < n) out[i] = f2bf(in[i]);
}

// zero-padded 2D convert: out[r, c] = (r < in_rows && c < in_cols) ? in : 0
__global__ void cvt_pad2d_kernel(const float* __restrict__ in,
                                 unsigned short* __restrict__ out,
                                 int in_rows, int in_cols,
                                 int out_rows, int out_cols) {
  int i = blockIdx.x * blockDim.x + threadIdx.x;
  if (i >= out_rows * out_cols) return;
  int r = i / out_cols, c = i % out_cols;
  float v = (r < in_rows && c < in_cols) ? in[(size_t)r * in_cols + c] : 0.0f;
  out[i] = f2bf(v);
}

__global__ void fill_u32_kernel(unsigned* __restrict__ p, unsigned v, int n) {
  int i = blockIdx.x * blockDim.x + threadIdx.x;
  if (i < n) p[i] = v;
}

__global__ void bias_act_cvt_kernel(const float* __restrict__ in,
                                    const float* __restrict__ bias,
                                    float* __restrict__ out,
                                    unsigned short* __restrict__ outbf,
                                    int total, int cols, float slope) {
  int i = blockIdx.x * blockDim.x + threadIdx.x;
  if (i >= total) return;
  int c = i % cols;
  float v = in[i] + bias[c];
  v = (v >= 0.0f) ? v : slope * v;
  out[i] = v;
  if (outbf) outbf[i] = f2bf(v);
}

// ---------------- WMMA GEMM: C[M,Nc] = act(A[M,K] @ B[K,Nc] + bias) ----------------
// Invariants: K % 32 == 0, Nc % 64 == 0. A,B bf16; C f32 (+ optional bf16 mirror).
// 256 threads = 8 waves. Block tile 128(M) x 64(N), K stepped by 32.
// A tile staged by the Tensor Data Mover (wave 0 issues tensor_load_to_lds,
// tensor_dim1 = M-m0 so out-of-range rows arrive as zeros). B tile staged by
// threads with one global_load_b128 + transposed ds_store_b16 scatter.
template<int ACT> // 0=none, 1=leaky(0.01), 2=relu
__global__ __launch_bounds__(256)
void gemm_bf16_kernel(const unsigned short* __restrict__ A,
                      const unsigned short* __restrict__ B,
                      const float* __restrict__ bias,
                      float* __restrict__ C,
                      unsigned short* __restrict__ Cbf,
                      int M, int K, int Nc) {
  __shared__ __align__(16) unsigned short As[128 * 32];
  __shared__ __align__(16) unsigned short Bs[64 * 32];

  const int tid  = threadIdx.x;
  const int lane = tid & 31;
  const int wv   = tid >> 5;
  const int half = lane >> 4;     // K-group selector per ISA 16-bit layout
  const int m0   = blockIdx.x * 128;
  const int n0   = blockIdx.y * 64;

  v8f acc[4];
  const v8f vzero = {0.f, 0.f, 0.f, 0.f, 0.f, 0.f, 0.f, 0.f};
#pragma unroll
  for (int t = 0; t < 4; ++t) acc[t] = vzero;

  // TDM descriptor pieces that are loop-invariant
  const unsigned ldsA   = (unsigned)(unsigned long long)(As); // addr[31:0] == LDS offset
  const unsigned tdim1  = (unsigned)(M - m0);                 // rows available (OOB -> 0)

  for (int k0 = 0; k0 < K; k0 += 32) {
    __syncthreads();   // previous tile fully consumed

    // ---- A tile [128 x 32] via Tensor Data Mover (async, wave 0 only) ----
    if (wv == 0) {
      unsigned long long ga =
          (unsigned long long)(A + (size_t)m0 * K + k0);
      uint32x4 g0;
      g0[0] = 1u;                                   // count=1, user mode
      g0[1] = ldsA;                                 // lds_addr
      g0[2] = (unsigned)(ga & 0xFFFFFFFFull);       // global_addr[31:0]
      g0[3] = (unsigned)((ga >> 32) & 0x1FFFFFFull) // global_addr[56:32]
              | (2u << 30);                         // type = 2 (image)
      int32x8 g1;
      g1[0] = 0x00010000;                           // data_size = 2 bytes
      g1[1] = (int)(32u << 16);                     // tensor_dim0[15:0]=32
      g1[2] = (int)((tdim1 & 0xFFFFu) << 16);       // tensor_dim0 hi=0 | tensor_dim1 lo
      g1[3] = (int)(((tdim1 >> 16) & 0xFFFFu)       // tensor_dim1 hi
              | (32u << 16));                       // tile_dim0 = 32
      g1[4] = 128;                                  // tile_dim1 = 128, tile_dim2 = 0
      g1[5] = K;                                    // tensor_dim0_stride[31:0]
      g1[6] = 0;
      g1[7] = 0;
      const int32x4 z4 = {0, 0, 0, 0};
      const int32x8 z8 = {0, 0, 0, 0, 0, 0, 0, 0};
      __builtin_amdgcn_tensor_load_to_lds(g0, g1, z4, z4, z8, 0);
    }

    // ---- B tile staged transposed: Bs[n][k] (unguarded: K%32==0, Nc%64==0) ----
    {
      int kk   = tid & 31;
      int nseg = tid >> 5;  // 0..7 -> 8 columns each
      union { uint4 q; unsigned short s[8]; } u;
      u.q = *(const uint4*)(B + (size_t)(k0 + kk) * Nc + n0 + nseg * 8);
#pragma unroll
      for (int j = 0; j < 8; ++j)
        Bs[(nseg * 8 + j) * 32 + kk] = u.s[j];
    }

    if (wv == 0) __builtin_amdgcn_s_wait_tensorcnt(0);
    __syncthreads();   // A (TDM) + B (ds) tiles visible to all waves

    // ---- fragments per ISA 16-bit A/B layout (16x32, K-pairs per dword) ----
    union FRG { v16bf v; unsigned u[8]; };
    FRG a;
    const unsigned short* arow = &As[(wv * 16 + (lane & 15)) * 32];
#pragma unroll
    for (int q = 0; q < 8; ++q) {
      int kb = ((q & 3) * 2) + ((q >> 2) * 16) + half * 8;
      a.u[q] = *(const unsigned*)(arow + kb);
    }
#pragma unroll
    for (int nt = 0; nt < 4; ++nt) {
      FRG b;
      const unsigned short* brow = &Bs[(nt * 16 + (lane & 15)) * 32];
#pragma unroll
      for (int q = 0; q < 8; ++q) {
        int kb = ((q & 3) * 2) + ((q >> 2) * 16) + half * 8;
        b.u[q] = *(const unsigned*)(brow + kb);
      }
      acc[nt] = __builtin_amdgcn_wmma_f32_16x16x32_bf16(
          false, a.v, false, b.v, (short)0, acc[nt], false, false);
    }
  }

  // ---- epilogue: C layout (lane<16: M=r, lane>=16: M=8+r; N=lane%16) ----
  const int col0  = n0 + (lane & 15);
  const int rbase = m0 + wv * 16 + half * 8;
#pragma unroll
  for (int nt = 0; nt < 4; ++nt) {
    int col  = col0 + nt * 16;
    float bb = bias ? bias[col] : 0.0f;
#pragma unroll
    for (int r = 0; r < 8; ++r) {
      int row = rbase + r;
      if (row < M) {
        float v = acc[nt][r] + bb;
        if (ACT == 1) v = (v >= 0.0f) ? v : 0.01f * v;
        if (ACT == 2) v = fmaxf(v, 0.0f);
        C[(size_t)row * Nc + col] = v;
        if (Cbf) Cbf[(size_t)row * Nc + col] = f2bf(v);
      }
    }
  }
}

// ---------------- GAT pieces ----------------
__global__ void gat_logits_kernel(const float* __restrict__ hw,
                                  const float* __restrict__ a_src,
                                  const float* __restrict__ a_dst,
                                  float* __restrict__ as_,
                                  float* __restrict__ ad_) {
  int t = blockIdx.x * blockDim.x + threadIdx.x;
  if (t >= N_NODES * H_HEADS) return;
  int node = t >> 2, h = t & 3;
  const float* base = hw + (size_t)node * HC + h * C_CH;
  const float* sw = a_src + h * C_CH;
  const float* dw = a_dst + h * C_CH;
  float accs = 0.f, accd = 0.f;
#pragma unroll 8
  for (int c = 0; c < C_CH; ++c) {
    float v = base[c];
    accs += v * sw[c];
    accd += v * dw[c];
  }
  as_[t] = accs;
  ad_[t] = accd;
}

__device__ __forceinline__ void edge_sd(const int* __restrict__ ei, int e,
                                        int& s, int& d) {
  if (e < E_EDGES) { s = ei[e]; d = ei[E_EDGES + e]; }
  else             { s = e - E_EDGES; d = s; }
}

__global__ void gat_edge_max_kernel(const int* __restrict__ ei,
                                    const float* __restrict__ as_,
                                    const float* __restrict__ ad_,
                                    unsigned* __restrict__ emaxKey) {
  int t = blockIdx.x * blockDim.x + threadIdx.x;
  if (t >= ETOT * H_HEADS) return;
  int e = t >> 2, h = t & 3;
  int s, d; edge_sd(ei, e, s, d);
  float x = as_[s * H_HEADS + h] + ad_[d * H_HEADS + h];
  x = (x >= 0.f) ? x : 0.2f * x;
  atomicMax(&emaxKey[d * H_HEADS + h], fkey(x));
}

__global__ void gat_edge_exp_kernel(const int* __restrict__ ei,
                                    const float* __restrict__ as_,
                                    const float* __restrict__ ad_,
                                    const unsigned* __restrict__ emaxKey,
                                    float* __restrict__ ee,
                                    float* __restrict__ denom) {
  int t = blockIdx.x * blockDim.x + threadIdx.x;
  if (t >= ETOT * H_HEADS) return;
  int e = t >> 2, h = t & 3;
  int s, d; edge_sd(ei, e, s, d);
  float x = as_[s * H_HEADS + h] + ad_[d * H_HEADS + h];
  x = (x >= 0.f) ? x : 0.2f * x;
  float m = fkey_inv(emaxKey[d * H_HEADS + h]);
  float v = __expf(x - m);
  ee[t] = v;
  atomicAdd(&denom[d * H_HEADS + h], v);
}

__global__ void gat_scatter_kernel(const int* __restrict__ ei,
                                   const float* __restrict__ hw,
                                   const float* __restrict__ ee,
                                   const float* __restrict__ denom,
                                   float* __restrict__ out) {
  int t = blockIdx.x * blockDim.x + threadIdx.x;
  if (t >= ETOT * 32) return;
  int e = t >> 5, p = t & 31;
  int c0 = p * 8;
  int h  = c0 >> 6;                 // C_CH == 64
  int s, d; edge_sd(ei, e, s, d);
  float alpha = ee[e * H_HEADS + h] / (denom[d * H_HEADS + h] + 1e-16f);
  const float* srcp = hw + (size_t)s * HC + c0;
  float* dstp = out + (size_t)d * HC + c0;
#pragma unroll
  for (int j = 0; j < 8; ++j) atomicAdd(&dstp[j], srcp[j] * alpha);
}

// ---------------- residual + LayerNorm (one wave32 per 256-wide row) ----------------
__global__ void add_ln_kernel(const float* __restrict__ x,
                              const float* __restrict__ y,
                              const float* __restrict__ g,
                              const float* __restrict__ b,
                              float* __restrict__ out,
                              unsigned short* __restrict__ outbf,
                              int rows) {
  int row  = blockIdx.x * 8 + (threadIdx.x >> 5);
  int lane = threadIdx.x & 31;
  if (row >= rows) return;
  const size_t base = (size_t)row * DM + lane * 8;
  float v[8]; float s = 0.f;
#pragma unroll
  for (int j = 0; j < 8; ++j) { v[j] = x[base + j] + y[base + j]; s += v[j]; }
#pragma unroll
  for (int off = 16; off > 0; off >>= 1) s += __shfl_xor(s, off, 32);
  float mean = s * (1.0f / DM);
  float var = 0.f;
#pragma unroll
  for (int j = 0; j < 8; ++j) { float dd = v[j] - mean; var += dd * dd; }
#pragma unroll
  for (int off = 16; off > 0; off >>= 1) var += __shfl_xor(var, off, 32);
  float inv = rsqrtf(var * (1.0f / DM) + 1e-5f);
  int c0 = lane * 8;
#pragma unroll
  for (int j = 0; j < 8; ++j) {
    float o = (v[j] - mean) * inv * g[c0 + j] + b[c0 + j];
    out[base + j] = o;
    if (outbf) outbf[base + j] = f2bf(o);
  }
}

// ---------------- global mean pool per graph ----------------
__global__ void pool_acc_kernel(const float* __restrict__ x,
                                const int* __restrict__ batch,
                                float* __restrict__ sums,
                                float* __restrict__ cnt) {
  int t = blockIdx.x * blockDim.x + threadIdx.x;
  if (t >= N_NODES * 32) return;
  int n = t >> 5, p = t & 31;
  int gidx = batch[n];
  const float* rowp = x + (size_t)n * DM + p * 8;
  float* sp = sums + (size_t)gidx * DM + p * 8;
#pragma unroll
  for (int j = 0; j < 8; ++j) atomicAdd(&sp[j], rowp[j]);
  if (p == 0) atomicAdd(&cnt[gidx], 1.0f);
}

__global__ void pool_fin_kernel(const float* __restrict__ sums,
                                const float* __restrict__ cnt,
                                unsigned short* __restrict__ pooledbf) {
  int i = blockIdx.x * blockDim.x + threadIdx.x;
  if (i >= G_GRAPHS * DM) return;
  float c = fmaxf(cnt[i / DM], 1.0f);
  pooledbf[i] = f2bf(sums[i] / c);
}

// ---------------- host orchestration ----------------
extern "C" void kernel_launch(void* const* d_in, const int* in_sizes, int n_in,
                              void* d_out, int out_size, void* d_ws, size_t ws_size,
                              hipStream_t stream) {
  (void)in_sizes; (void)n_in; (void)out_size; (void)ws_size;

  const float* x      = (const float*)d_in[0];
  const int*   ei     = (const int*)d_in[1];
  const int*   batch  = (const int*)d_in[2];
  const float* featW  = (const float*)d_in[3];
  const float* featB  = (const float*)d_in[4];
  const float* g1W    = (const float*)d_in[5];
  const float* g1as   = (const float*)d_in[6];
  const float* g1ad   = (const float*)d_in[7];
  const float* g1b    = (const float*)d_in[8];
  const float* g2W    = (const float*)d_in[9];
  const float* g2as   = (const float*)d_in[10];
  const float* g2ad   = (const float*)d_in[11];
  const float* g2b    = (const float*)d_in[12];
  // enc.attn: Wq=13 bq=14 Wk=15 bk=16 Wv=17 bv=18 Wo=19 bo=20 (Wq/Wk dead for seq=1)
  const float* encWv  = (const float*)d_in[17];
  const float* encbv  = (const float*)d_in[18];
  const float* encWo  = (const float*)d_in[19];
  const float* encbo  = (const float*)d_in[20];
  const float* eln1g  = (const float*)d_in[21];
  const float* eln1b  = (const float*)d_in[22];
  const float* encW1  = (const float*)d_in[23];
  const float* encb1  = (const float*)d_in[24];
  const float* encW2  = (const float*)d_in[25];
  const float* encb2  = (const float*)d_in[26];
  const float* eln2g  = (const float*)d_in[27];
  const float* eln2b  = (const float*)d_in[28];
  // dec.self: 29..36
  const float* dsWv   = (const float*)d_in[33];
  const float* dsbv   = (const float*)d_in[34];
  const float* dsWo   = (const float*)d_in[35];
  const float* dsbo   = (const float*)d_in[36];
  const float* dln1g  = (const float*)d_in[37];
  const float* dln1b  = (const float*)d_in[38];
  // dec.cross: 39..46
  const float* dcWv   = (const float*)d_in[43];
  const float* dcbv   = (const float*)d_in[44];
  const float* dcWo   = (const float*)d_in[45];
  const float* dcbo   = (const float*)d_in[46];
  const float* dln2g  = (const float*)d_in[47];
  const float* dln2b  = (const float*)d_in[48];
  const float* dW1    = (const float*)d_in[49];
  const float* db1    = (const float*)d_in[50];
  const float* dW2    = (const float*)d_in[51];
  const float* db2    = (const float*)d_in[52];
  const float* dln3g  = (const float*)d_in[53];
  const float* dln3b  = (const float*)d_in[54];
  const float* fcW    = (const float*)d_in[55];
  const float* fcb    = (const float*)d_in[56];

  char* wsp = (char*)d_ws;
  size_t off = 0;
  auto alloc = [&](size_t bytes) -> void* {
    off = (off + 255) & ~(size_t)255;
    void* p = wsp + off;
    off += bytes;
    return p;
  };

  // bf16 weights / padded inputs
  unsigned short* xbf     = (unsigned short*)alloc((size_t)N_NODES * F_PAD * 2);
  unsigned short* featWbf = (unsigned short*)alloc((size_t)F_PAD * D_FEAT * 2);
  unsigned short* g1Wbf   = (unsigned short*)alloc((size_t)D_FEAT * HC * 2);
  unsigned short* g2Wbf   = (unsigned short*)alloc((size_t)HC * HC * 2);
  unsigned short* encWvbf = (unsigned short*)alloc((size_t)DM * DM * 2);
  unsigned short* encWobf = (unsigned short*)alloc((size_t)DM * DM * 2);
  unsigned short* dsWvbf  = (unsigned short*)alloc((size_t)DM * DM * 2);
  unsigned short* dsWobf  = (unsigned short*)alloc((size_t)DM * DM * 2);
  unsigned short* dcWvbf  = (unsigned short*)alloc((size_t)DM * DM * 2);
  unsigned short* dcWobf  = (unsigned short*)alloc((size_t)DM * DM * 2);
  unsigned short* encW1bf = (unsigned short*)alloc((size_t)DM * FFD * 2);
  unsigned short* encW2bf = (unsigned short*)alloc((size_t)FFD * DM * 2);
  unsigned short* dW1bf   = (unsigned short*)alloc((size_t)DM * FFD * 2);
  unsigned short* dW2bf   = (unsigned short*)alloc((size_t)FFD * DM * 2);
  unsigned short* fcWbf   = (unsigned short*)alloc((size_t)DM * OUTD * 2);

  // activations
  float*          h64   = (float*)alloc((size_t)N_NODES * D_FEAT * 4);
  unsigned short* h64bf = (unsigned short*)alloc((size_t)N_NODES * D_FEAT * 2);
  float*          hw    = (float*)alloc((size_t)N_NODES * HC * 4);   // gat proj (reused)
  float*          as_   = (float*)alloc((size_t)N_NODES * H_HEADS * 4);
  float*          ad_   = (float*)alloc((size_t)N_NODES * H_HEADS * 4);
  unsigned*       emaxK = (unsigned*)alloc((size_t)N_NODES * H_HEADS * 4);
  float*          denom = (float*)alloc((size_t)N_NODES * H_HEADS * 4);
  float*          ee    = (float*)alloc((size_t)ETOT * H_HEADS * 4);
  float*          gacc  = (float*)alloc((size_t)N_NODES * HC * 4);
  float*          g1f   = (float*)alloc((size_t)N_NODES * HC * 4);
  unsigned short* g1bf  = (unsigned short*)alloc((size_t)N_NODES * HC * 2);
  float*          h2f   = (float*)alloc((size_t)N_NODES * HC * 4);
  unsigned short* h2bf  = (unsigned short*)alloc((size_t)N_NODES * HC * 2);
  float*          tv    = (float*)alloc((size_t)N_NODES * DM * 4);
  unsigned short* tvbf  = (unsigned short*)alloc((size_t)N_NODES * DM * 2);
  float*          to    = (float*)alloc((size_t)N_NODES * DM * 4);
  float*          memf  = (float*)alloc((size_t)N_NODES * DM * 4);
  unsigned short* membf = (unsigned short*)alloc((size_t)N_NODES * DM * 2);
  float*          f1    = (float*)alloc((size_t)N_NODES * FFD * 4);
  unsigned short* f1bf  = (unsigned short*)alloc((size_t)N_NODES * FFD * 2);
  float*          f2    = (float*)alloc((size_t)N_NODES * DM * 4);
  float*          mem2f = (float*)alloc((size_t)N_NODES * DM * 4);
  unsigned short* mem2bf= (unsigned short*)alloc((size_t)N_NODES * DM * 2);
  float*          d1f   = (float*)alloc((size_t)N_NODES * DM * 4);
  float*          d2f   = (float*)alloc((size_t)N_NODES * DM * 4);
  unsigned short* d2bf  = (unsigned short*)alloc((size_t)N_NODES * DM * 2);
  float*          d3f   = (float*)alloc((size_t)N_NODES * DM * 4);
  float*          sums  = (float*)alloc((size_t)G_GRAPHS * DM * 4);
  float*          cnt   = (float*)alloc((size_t)G_GRAPHS * 4);
  unsigned short* pbf   = (unsigned short*)alloc((size_t)G_GRAPHS * DM * 2);

  auto cvt = [&](const float* s, unsigned short* dst, int n) {
    cvt_bf16_kernel<<<(n + 255) / 256, 256, 0, stream>>>(s, dst, n);
  };
  auto fill = [&](void* p, unsigned v, int n) {
    fill_u32_kernel<<<(n + 255) / 256, 256, 0, stream>>>((unsigned*)p, v, n);
  };
  auto gemm = [&](const unsigned short* A, const unsigned short* B, const float* bias,
                  float* C, unsigned short* Cbf, int M, int K, int Nc, int act) {
    dim3 grid((M + 127) / 128, Nc / 64);
    if (act == 1)
      gemm_bf16_kernel<1><<<grid, 256, 0, stream>>>(A, B, bias, C, Cbf, M, K, Nc);
    else if (act == 2)
      gemm_bf16_kernel<2><<<grid, 256, 0, stream>>>(A, B, bias, C, Cbf, M, K, Nc);
    else
      gemm_bf16_kernel<0><<<grid, 256, 0, stream>>>(A, B, bias, C, Cbf, M, K, Nc);
  };
  auto gat_layer = [&](const float* hwin, const float* aw_s, const float* aw_d,
                       const float* biasv, float* outf, unsigned short* outbf) {
    fill(emaxK, 0u, N_NODES * H_HEADS);
    fill(denom, 0u, N_NODES * H_HEADS);
    fill(gacc, 0u, N_NODES * HC);
    gat_logits_kernel<<<(N_NODES * H_HEADS + 255) / 256, 256, 0, stream>>>(
        hwin, aw_s, aw_d, as_, ad_);
    gat_edge_max_kernel<<<(ETOT * H_HEADS + 255) / 256, 256, 0, stream>>>(
        ei, as_, ad_, emaxK);
    gat_edge_exp_kernel<<<(ETOT * H_HEADS + 255) / 256, 256, 0, stream>>>(
        ei, as_, ad_, emaxK, ee, denom);
    gat_scatter_kernel<<<(ETOT * 32 + 255) / 256, 256, 0, stream>>>(
        ei, hwin, ee, denom, gacc);
    bias_act_cvt_kernel<<<(N_NODES * HC + 255) / 256, 256, 0, stream>>>(
        gacc, biasv, outf, outbf, N_NODES * HC, HC, 0.01f);
  };
  auto lnres = [&](const float* xr, const float* yr, const float* g, const float* b,
                   float* o, unsigned short* obf) {
    add_ln_kernel<<<(N_NODES + 7) / 8, 256, 0, stream>>>(xr, yr, g, b, o, obf, N_NODES);
  };

  // ---- weight / input conversion (feat K padded 16 -> 32) ----
  cvt_pad2d_kernel<<<(N_NODES * F_PAD + 255) / 256, 256, 0, stream>>>(
      x, xbf, N_NODES, F_IN, N_NODES, F_PAD);
  cvt_pad2d_kernel<<<(F_PAD * D_FEAT + 255) / 256, 256, 0, stream>>>(
      featW, featWbf, F_IN, D_FEAT, F_PAD, D_FEAT);
  cvt(g1W, g1Wbf, D_FEAT * HC);
  cvt(g2W, g2Wbf, HC * HC);
  cvt(encWv, encWvbf, DM * DM);
  cvt(encWo, encWobf, DM * DM);
  cvt(dsWv, dsWvbf, DM * DM);
  cvt(dsWo, dsWobf, DM * DM);
  cvt(dcWv, dcWvbf, DM * DM);
  cvt(dcWo, dcWobf, DM * DM);
  cvt(encW1, encW1bf, DM * FFD);
  cvt(encW2, encW2bf, FFD * DM);
  cvt(dW1, dW1bf, DM * FFD);
  cvt(dW2, dW2bf, FFD * DM);
  cvt(fcW, fcWbf, DM * OUTD);

  // ---- feature projection (K padded to 32) ----
  gemm(xbf, featWbf, featB, h64, h64bf, N_NODES, F_PAD, D_FEAT, 0);

  // ---- GAT layer 1 ----
  gemm(h64bf, g1Wbf, nullptr, hw, nullptr, N_NODES, D_FEAT, HC, 0);
  gat_layer(hw, g1as, g1ad, g1b, g1f, g1bf);

  // ---- GAT layer 2 ----
  gemm(g1bf, g2Wbf, nullptr, hw, nullptr, N_NODES, HC, HC, 0);
  gat_layer(hw, g2as, g2ad, g2b, h2f, h2bf);

  // ---- encoder: attn(seq=1) == V path only, then post-norm + FFN ----
  gemm(h2bf, encWvbf, encbv, tv, tvbf, N_NODES, DM, DM, 0);
  gemm(tvbf, encWobf, encbo, to, nullptr, N_NODES, DM, DM, 0);
  lnres(h2f, to, eln1g, eln1b, memf, membf);
  gemm(membf, encW1bf, encb1, f1, f1bf, N_NODES, DM, FFD, 2);
  gemm(f1bf, encW2bf, encb2, f2, nullptr, N_NODES, FFD, DM, 0);
  lnres(memf, f2, eln2g, eln2b, mem2f, mem2bf);

  // ---- decoder ----
  gemm(h2bf, dsWvbf, dsbv, tv, tvbf, N_NODES, DM, DM, 0);
  gemm(tvbf, dsWobf, dsbo, to, nullptr, N_NODES, DM, DM, 0);
  lnres(h2f, to, dln1g, dln1b, d1f, nullptr);
  gemm(mem2bf, dcWvbf, dcbv, tv, tvbf, N_NODES, DM, DM, 0);
  gemm(tvbf, dcWobf, dcbo, to, nullptr, N_NODES, DM, DM, 0);
  lnres(d1f, to, dln2g, dln2b, d2f, d2bf);
  gemm(d2bf, dW1bf, db1, f1, f1bf, N_NODES, DM, FFD, 2);
  gemm(f1bf, dW2bf, db2, f2, nullptr, N_NODES, FFD, DM, 0);
  lnres(d2f, f2, dln3g, dln3b, d3f, nullptr);

  // ---- global mean pool + fc ----
  fill(sums, 0u, G_GRAPHS * DM);
  fill(cnt, 0u, G_GRAPHS);
  pool_acc_kernel<<<(N_NODES * 32 + 255) / 256, 256, 0, stream>>>(d3f, batch, sums, cnt);
  pool_fin_kernel<<<(G_GRAPHS * DM + 255) / 256, 256, 0, stream>>>(sums, cnt, pbf);
  gemm(pbf, fcWbf, fcb, (float*)d_out, nullptr, G_GRAPHS, DM, OUTD, 0);
}
